// ToyMultiHeadAttention_11854109737050
// MI455X (gfx1250) — matrix operand order repaired and checked by reading, not compiled, not run on hardware
//
#include <hip/hip_runtime.h>
#include <hip/hip_bf16.h>

#define B_ 4
#define T_ 2048
#define C_ 2048
#define H_ 16
#define D_ 128

typedef __bf16 bf16_t;
typedef __attribute__((ext_vector_type(16))) __bf16 v16bf;
typedef __attribute__((ext_vector_type(8)))  __bf16 v8bf;
typedef __attribute__((ext_vector_type(8)))  float   v8f;

union ABFrag { v16bf v; v8bf h[2]; };

static __device__ __forceinline__ v8f wmma_bf16(const ABFrag& a, const ABFrag& b, v8f c) {
  return __builtin_amdgcn_wmma_f32_16x16x32_bf16(false, a.v, false, b.v,
                                                 (short)0, c, false, false);
}

// A fragment: row m, element i -> K = (i<8 ? i : i+8) + 8*kh
static __device__ __forceinline__ void load_afrag(ABFrag& a, const bf16_t* Xrow,
                                                  int k0, int kh) {
  a.h[0] = *(const v8bf*)(Xrow + k0 + 8 * kh);
  a.h[1] = *(const v8bf*)(Xrow + k0 + 16 + 8 * kh);
}

// B fragment: col m, element i -> K = (i<8 ? i : i+8) + 8*kh
static __device__ __forceinline__ void load_bfrag(ABFrag& b, const bf16_t* Wrow,
                                                  int k0, int kh) {
  b.h[0] = *(const v8bf*)(Wrow + k0 + 8 * kh);
  b.h[1] = *(const v8bf*)(Wrow + k0 + 16 + 8 * kh);
}

// ---------------------------------------------------------------------------
// fp32 -> bf16 conversion (grid-stride)
// ---------------------------------------------------------------------------
__global__ __launch_bounds__(256) void cvt_f32_to_bf16(const float* __restrict__ in,
                                                       bf16_t* __restrict__ out, int n) {
  int i = blockIdx.x * blockDim.x + threadIdx.x;
  const int stride = gridDim.x * blockDim.x;
  for (; i < n; i += stride) out[i] = (bf16_t)in[i];
}

// ---------------------------------------------------------------------------
// GEMM:  Y[row, n] = sum_k X[row,k] * W[n,k] + bias[n]
// Wave tile 32 x 64: 2 A-frags + 4 B-frags feed 8 WMMAs per 32-K step
// (A reused x4, B reused x2).  Plain single-step K loop: the backend's
// software pipeliner hoists the next step's loads across the WMMA chain
// on its own (partial s_wait_loadcnt) without needing accumulator
// phase-copies (which incur WMMA->VALU hazard NOPs).
// Block: 256 threads = 8 waves (2 M-waves x 4 N-waves) -> block tile 64 x 256.
// mode 0: bf16 [B,H,T,D] | mode 1: bf16 [B,H,D,T] | mode 2: fp32 [B,T,C]
// ---------------------------------------------------------------------------
__global__ __launch_bounds__(256) void gemm_proj(
    const bf16_t* __restrict__ X, const bf16_t* __restrict__ W,
    const float* __restrict__ bias, bf16_t* __restrict__ out_bf,
    float* __restrict__ out_f, int mode)
{
  const int tid  = threadIdx.x;
  const int lane = tid & 31;
  const int wid  = tid >> 5;
  const int wm   = wid & 1;      // 2 M-waves
  const int wn   = wid >> 1;     // 4 N-waves
  const int m    = lane & 15;
  const int kh   = lane >> 4;

  const int row0 = blockIdx.x * 64 + wm * 32;
  const int n0   = blockIdx.y * 256 + wn * 64;

  const bf16_t* Xrow0 = X + (size_t)(row0 + m) * C_;
  const bf16_t* Xrow1 = X + (size_t)(row0 + 16 + m) * C_;
  const bf16_t* Wr[4];
#pragma unroll
  for (int t = 0; t < 4; ++t)
    Wr[t] = W + (size_t)(n0 + 16 * t + m) * C_;

  v8f acc[2][4];
#pragma unroll
  for (int s = 0; s < 2; ++s)
#pragma unroll
    for (int t = 0; t < 4; ++t)
#pragma unroll
      for (int e = 0; e < 8; ++e) acc[s][t][e] = 0.0f;

  for (int k0 = 0; k0 < C_; k0 += 32) {
    // all fragment loads for this K step (one clause), then 8 WMMAs
    ABFrag a0, a1, b[4];
    load_afrag(a0, Xrow0, k0, kh);
    load_afrag(a1, Xrow1, k0, kh);
#pragma unroll
    for (int t = 0; t < 4; ++t) load_bfrag(b[t], Wr[t], k0, kh);
    __builtin_prefetch(Xrow0 + k0 + 96, 0, 3);

#pragma unroll
    for (int t = 0; t < 4; ++t) {
      acc[0][t] = wmma_bf16(a0, b[t], acc[0][t]);
      acc[1][t] = wmma_bf16(a1, b[t], acc[1][t]);
    }
  }

  // C/D layout: vgpr r, lane l -> row = r + 8*(l>>4), col = l&15
#pragma unroll
  for (int s = 0; s < 2; ++s) {
#pragma unroll
    for (int t = 0; t < 4; ++t) {
      const int n  = n0 + 16 * t + m;
      const float bv = bias[n];
#pragma unroll
      for (int r = 0; r < 8; ++r) {
        const int row = row0 + 16 * s + r + 8 * kh;
        const float y = acc[s][t][r] + bv;
        if (mode == 0) {
          const int bb = row / T_, tt = row % T_;
          const int hh = n / D_,  dd = n % D_;
          out_bf[(((size_t)bb * H_ + hh) * T_ + tt) * D_ + dd] = (bf16_t)y;
        } else if (mode == 1) {
          const int bb = row / T_, tt = row % T_;
          const int hh = n / D_,  dd = n % D_;
          out_bf[(((size_t)bb * H_ + hh) * D_ + dd) * T_ + tt] = (bf16_t)y;
        } else {
          out_f[(size_t)row * C_ + n] = y;
        }
      }
    }
  }
}

// ---------------------------------------------------------------------------
// Flash attention.  One wave per 16-row Q tile; 4 waves (64 Q rows) per block.
// Q,K: [B,H,T,D] bf16.  V: [B,H,D,T] bf16 (transposed).  Out: [B,T,C] bf16.
// All K and V fragments for a 32-key step are loaded up-front, so V-load
// latency hides under the S WMMAs + softmax VALU/trans work.
// ---------------------------------------------------------------------------
__global__ __launch_bounds__(128) void flash_attn(
    const bf16_t* __restrict__ Qh, const bf16_t* __restrict__ Kh,
    const bf16_t* __restrict__ Vt, bf16_t* __restrict__ attn)
{
  __shared__ __align__(16) bf16_t pbuf[4][16][32];

  const int tid  = threadIdx.x;
  const int lane = tid & 31;
  const int w    = tid >> 5;
  const int m    = lane & 15;
  const int kh   = lane >> 4;

  const int h = blockIdx.y;
  const int b = blockIdx.z;
  const int qbase = blockIdx.x * 64 + w * 16;
  const int qend  = qbase + 15;

  const size_t headQK = ((size_t)b * H_ + h) * (size_t)T_ * D_;
  const size_t headV  = ((size_t)b * H_ + h) * (size_t)D_ * T_;

  // Preload Q tile (16 x 128) as 4 A-fragments of 16x32
  ABFrag qa[4];
  {
    const bf16_t* Qrow = Qh + headQK + (size_t)(qbase + m) * D_;
#pragma unroll
    for (int kk = 0; kk < 4; ++kk) load_afrag(qa[kk], Qrow, 32 * kk, kh);
  }

  v8f o[8];
#pragma unroll
  for (int c = 0; c < 8; ++c)
#pragma unroll
    for (int e = 0; e < 8; ++e) o[c][e] = 0.0f;

  float mrow[8], lrow[8];
#pragma unroll
  for (int r = 0; r < 8; ++r) { mrow[r] = -1e30f; lrow[r] = 0.0f; }

  const float scale = 0.08838834764831845f;  // 1/sqrt(128)

  for (int jb = 0; jb <= qend; jb += 32) {
    // ---- issue ALL loads for this 32-key step up front ----
    const bf16_t* Krow0 = Kh + headQK + (size_t)(jb + m) * D_;
    const bf16_t* Krow1 = Kh + headQK + (size_t)(jb + 16 + m) * D_;
    ABFrag kb[8];
#pragma unroll
    for (int kk = 0; kk < 4; ++kk) {
      load_bfrag(kb[2 * kk],     Krow0, 32 * kk, kh);
      load_bfrag(kb[2 * kk + 1], Krow1, 32 * kk, kh);
    }
    ABFrag vb[8];
#pragma unroll
    for (int c = 0; c < 8; ++c) {
      const bf16_t* Vrow = Vt + headV + (size_t)(16 * c + m) * T_;
      load_bfrag(vb[c], Vrow, jb, kh);
    }

    // ---- S = Q . K^T for 32 keys (two 16-col accumulators) ----
    v8f sa, sb;
#pragma unroll
    for (int e = 0; e < 8; ++e) { sa[e] = 0.0f; sb[e] = 0.0f; }
#pragma unroll
    for (int kk = 0; kk < 4; ++kk) {
      sa = wmma_bf16(qa[kk], kb[2 * kk],     sa);
      sb = wmma_bf16(qa[kk], kb[2 * kk + 1], sb);
    }

    // ---- online softmax, per output row (r + 8*kh) ----
#pragma unroll
    for (int r = 0; r < 8; ++r) {
      const int qrow = qbase + r + 8 * kh;
      float sA = sa[r] * scale;
      float sB = sb[r] * scale;
      if (jb + m > qrow)      sA = -1e30f;   // causal mask
      if (jb + 16 + m > qrow) sB = -1e30f;
      float vmax = fmaxf(sA, sB);
#pragma unroll
      for (int off = 8; off > 0; off >>= 1)
        vmax = fmaxf(vmax, __shfl_xor(vmax, off, 32));
      const float mnew = fmaxf(mrow[r], vmax);
      const float fac  = __expf(mrow[r] - mnew);
      const float pa   = __expf(sA - mnew);
      const float pb   = __expf(sB - mnew);
      float rs = pa + pb;
#pragma unroll
      for (int off = 8; off > 0; off >>= 1)
        rs += __shfl_xor(rs, off, 32);
      lrow[r] = lrow[r] * fac + rs;
      mrow[r] = mnew;
#pragma unroll
      for (int c = 0; c < 8; ++c) o[c][r] *= fac;
      pbuf[w][r + 8 * kh][m]      = (bf16_t)pa;
      pbuf[w][r + 8 * kh][16 + m] = (bf16_t)pb;
    }

    // same-wave DS ordering: drain LDS stores before the repack loads
    asm volatile("s_wait_dscnt 0" ::: "memory");

    // repack P (C-layout) -> A-fragment layout via LDS
    ABFrag pf;
    pf.h[0] = *(const v8bf*)(&pbuf[w][m][8 * kh]);
    pf.h[1] = *(const v8bf*)(&pbuf[w][m][16 + 8 * kh]);

    // ---- O += P . V ----
#pragma unroll
    for (int c = 0; c < 8; ++c)
      o[c] = wmma_bf16(pf, vb[c], o[c]);
  }

  // ---- normalize and store [B,T,C] bf16 ----
#pragma unroll
  for (int r = 0; r < 8; ++r) {
    const int qrow = qbase + r + 8 * kh;
    const float inv = 1.0f / lrow[r];
    bf16_t* orow = attn + ((size_t)b * T_ + qrow) * C_ + h * D_;
#pragma unroll
    for (int c = 0; c < 8; ++c)
      orow[16 * c + m] = (bf16_t)(o[c][r] * inv);
  }
}

// ---------------------------------------------------------------------------
extern "C" void kernel_launch(void* const* d_in, const int* in_sizes, int n_in,
                              void* d_out, int out_size, void* d_ws, size_t ws_size,
                              hipStream_t stream) {
  const float* q  = (const float*)d_in[0];
  const float* k  = (const float*)d_in[1];
  const float* v  = (const float*)d_in[2];
  const float* Wq = (const float*)d_in[3];
  const float* bq = (const float*)d_in[4];
  const float* Wk = (const float*)d_in[5];
  const float* bk = (const float*)d_in[6];
  const float* Wv = (const float*)d_in[7];
  const float* bv = (const float*)d_in[8];
  const float* Wo = (const float*)d_in[9];
  const float* bo = (const float*)d_in[10];
  float* out = (float*)d_out;

  char* ws = (char*)d_ws;
  const int nBTC = B_ * T_ * C_;   // 16,777,216
  const int nCC  = C_ * C_;        //  4,194,304

  // workspace map (192 MB total)
  bf16_t* Xbuf = (bf16_t*)(ws);                    // 32 MB (reused per proj)
  bf16_t* Wqb  = (bf16_t*)(ws + (32ull  << 20));   //  8 MB
  bf16_t* Wkb  = (bf16_t*)(ws + (40ull  << 20));
  bf16_t* Wvb  = (bf16_t*)(ws + (48ull  << 20));
  bf16_t* Wob  = (bf16_t*)(ws + (56ull  << 20));
  bf16_t* Qh   = (bf16_t*)(ws + (64ull  << 20));   // 32 MB [B,H,T,D]
  bf16_t* Khd  = (bf16_t*)(ws + (96ull  << 20));   // 32 MB [B,H,T,D]
  bf16_t* Vt   = (bf16_t*)(ws + (128ull << 20));   // 32 MB [B,H,D,T]
  bf16_t* attn = (bf16_t*)(ws + (160ull << 20));   // 32 MB [B,T,C]

  const dim3 cg(2048), cb(256);
  const dim3 gg((B_ * T_) / 64, C_ / 256), gb(256);

  cvt_f32_to_bf16<<<cg, cb, 0, stream>>>(Wq, Wqb, nCC);
  cvt_f32_to_bf16<<<cg, cb, 0, stream>>>(Wk, Wkb, nCC);
  cvt_f32_to_bf16<<<cg, cb, 0, stream>>>(Wv, Wvb, nCC);
  cvt_f32_to_bf16<<<cg, cb, 0, stream>>>(Wo, Wob, nCC);

  cvt_f32_to_bf16<<<cg, cb, 0, stream>>>(q, Xbuf, nBTC);
  gemm_proj<<<gg, gb, 0, stream>>>(Xbuf, Wqb, bq, Qh, nullptr, 0);

  cvt_f32_to_bf16<<<cg, cb, 0, stream>>>(k, Xbuf, nBTC);
  gemm_proj<<<gg, gb, 0, stream>>>(Xbuf, Wkb, bk, Khd, nullptr, 0);

  cvt_f32_to_bf16<<<cg, cb, 0, stream>>>(v, Xbuf, nBTC);
  gemm_proj<<<gg, gb, 0, stream>>>(Xbuf, Wvb, bv, Vt, nullptr, 1);

  flash_attn<<<dim3(T_ / 64, H_, B_), 128, 0, stream>>>(Qh, Khd, Vt, attn);

  gemm_proj<<<gg, gb, 0, stream>>>(attn, Wob, bo, nullptr, out, 2);

  (void)in_sizes; (void)n_in; (void)out_size; (void)ws_size;
}
